// MultiHeadAttention_14147622273688
// MI455X (gfx1250) — matrix operand order repaired and checked by reading, not compiled, not run on hardware
//
#include <hip/hip_runtime.h>
#include <hip/hip_fp16.h>

#define H  8
#define DK 32
#define DM 256
#define NQ 16384
#define NK 262144

typedef __attribute__((ext_vector_type(16))) _Float16 v16h;
typedef __attribute__((ext_vector_type(8)))  float    v8f;
typedef __attribute__((ext_vector_type(4)))  unsigned int u32x4;
typedef __attribute__((ext_vector_type(8)))  int i32x8;
typedef __attribute__((ext_vector_type(4)))  int i32x4;

#define WP 264   // LDS row pitch in halfs: 256 data + 8 pad = 528B rows (TDM pad: 4 DWORDs per 128 DWORDs)

// fp32 -> f16 conversion of the 4 weight matrices (Wq,Wk,Wv,Wo), row-major 256x256 each.
__global__ void cvt_w_kernel(const float* __restrict__ W0, const float* __restrict__ W1,
                             const float* __restrict__ W2, const float* __restrict__ W3,
                             _Float16* __restrict__ dst) {
  const int c = blockIdx.x*256 + threadIdx.x;       // 8-element chunk id, 32768 total
  const int m = c >> 13;                            // matrix index (8192 chunks each)
  const int o = (c & 8191) << 3;
  const float* src = (m == 0) ? W0 : (m == 1) ? W1 : (m == 2) ? W2 : W3;
  const float4 f0 = *(const float4*)(src + o);
  const float4 f1 = *(const float4*)(src + o + 4);
  union { uint4 u; _Float16 h[8]; } pk;
  pk.h[0]=(_Float16)f0.x; pk.h[1]=(_Float16)f0.y; pk.h[2]=(_Float16)f0.z; pk.h[3]=(_Float16)f0.w;
  pk.h[4]=(_Float16)f1.x; pk.h[5]=(_Float16)f1.y; pk.h[6]=(_Float16)f1.z; pk.h[7]=(_Float16)f1.w;
  *(uint4*)(dst + (size_t)m*65536 + o) = pk.u;
}

// C[M x 256] = A[M x 256] @ Wh[256 x 256] + bias, f16 WMMA, f32 accum.
// Weight (pre-converted f16) staged into LDS by the Tensor Data Mover with HW row padding.
// EPI==0: store f16 activations. EPI==1: += residual, BatchNorm(eval), store f32.
template<int EPI>
__global__ __launch_bounds__(256) void gemm256_kernel(
    const float* __restrict__ A, const _Float16* __restrict__ Wh,
    const float* __restrict__ bias,
    const float* __restrict__ resid,
    const float* __restrict__ gamma, const float* __restrict__ beta,
    const float* __restrict__ mean,  const float* __restrict__ var,
    _Float16* __restrict__ outh, float* __restrict__ outf)
{
  extern __shared__ _Float16 Wlds[];   // 256 x WP halfs = 132 KB (gfx1250 WGP LDS = 320 KB)
  const int tid = threadIdx.x;

  // --- TDM: DMA the whole 256x256 f16 weight into LDS, padding 16B after every 512B row ---
  if (tid < 32) {                       // wave 0 only: TDM issue is per-wave, EXEC-independent
    const unsigned int ldsbase = __builtin_amdgcn_groupstaticsize();  // dynamic LDS starts here (0)
    const unsigned long long ga = (unsigned long long)(uintptr_t)Wh;
    u32x4 g0;
    g0[0] = 1u;                                            // count=1, user mode, no gather
    g0[1] = ldsbase;                                       // lds_addr (bytes)
    g0[2] = (unsigned int)ga;                              // global_addr[31:0]
    g0[3] = (unsigned int)((ga >> 32) & 0x01FFFFFFu)       // global_addr[56:32]
          | 0x80000000u;                                   // type=2 ("image")
    i32x8 g1;
    g1[0] = (int)((1u << 16)      // data_size = 1 (2 bytes)
                | (1u << 20)      // pad_enable
                | (6u << 22)      // pad_interval: every 128 DWORDs (= one 512B row)
                | (3u << 25));    // pad_amount: 4 DWORDs (16B) -> 264-half pitch
    g1[1] = (int)(256u << 16);    // tensor_dim0[15:0] in [31:16]
    g1[2] = (int)(256u << 16);    // tensor_dim1[15:0] in [31:16] (dim0 hi = 0)
    g1[3] = (int)(256u << 16);    // tile_dim0 in [31:16] (dim1 hi = 0)
    g1[4] = 256;                  // tile_dim1 (tile_dim2 = 0 -> 2D)
    g1[5] = 256;                  // tensor_dim0_stride[31:0]
    g1[6] = 0;
    g1[7] = 0;
    const i32x4 z4 = {0, 0, 0, 0};
    const i32x8 z8 = {0, 0, 0, 0, 0, 0, 0, 0};
    __builtin_amdgcn_tensor_load_to_lds(g0, g1, z4, z4, z8, 0);
    __builtin_amdgcn_s_wait_tensorcnt(0);
  }
  __syncthreads();

  const int wave = tid >> 5, lane = tid & 31;
  const int hsel = lane >> 4, nl = lane & 15;
  const int mrow = blockIdx.x*128 + wave*16 + nl;   // A-frag row (lanes 0-15 == rows 0-15, repeated)

  v8f acc[16];
  #pragma unroll
  for (int nt = 0; nt < 16; ++nt) {
    const float b = bias[nt*16 + nl];               // C/D layout: N = lane%16 (+tile), M = vgpr + 8*(lane/16)
    #pragma unroll
    for (int r = 0; r < 8; ++r) acc[nt][r] = b;
  }

  for (int kk = 0; kk < 8; ++kk) {                  // K = 256 in steps of 32
    // 16-bit A 16x32 layout: lane-half h -> elems 0..7 = K kk*32+8h.., elems 8..15 = K kk*32+16+8h..
    const float* ar = A + (size_t)mrow*256 + kk*32 + hsel*8;
    const float4 a0 = *(const float4*)(ar);
    const float4 a1 = *(const float4*)(ar + 4);
    const float4 a2 = *(const float4*)(ar + 16);
    const float4 a3 = *(const float4*)(ar + 20);
    v16h af;
    af[0]=(_Float16)a0.x; af[1]=(_Float16)a0.y; af[2]=(_Float16)a0.z; af[3]=(_Float16)a0.w;
    af[4]=(_Float16)a1.x; af[5]=(_Float16)a1.y; af[6]=(_Float16)a1.z; af[7]=(_Float16)a1.w;
    af[8]=(_Float16)a2.x; af[9]=(_Float16)a2.y; af[10]=(_Float16)a2.z; af[11]=(_Float16)a2.w;
    af[12]=(_Float16)a3.x; af[13]=(_Float16)a3.y; af[14]=(_Float16)a3.z; af[15]=(_Float16)a3.w;

    // 16-bit B 32x16 layout: N = lane%16, lane-half h -> elems 0..7 = K kk*32+8h.., elems 8..15 = K kk*32+16+8h..
    const _Float16* brow = Wlds + (kk*32 + hsel*8)*WP + nl;
    #pragma unroll
    for (int nt = 0; nt < 16; ++nt) {
      union { uint4 u[2]; v16h v; } bf;
      #pragma unroll
      for (int s = 0; s < 8; ++s) {
        bf.v[s]     = brow[s*WP + nt*16];
        bf.v[s + 8] = brow[(16 + s)*WP + nt*16];
      }
      acc[nt] = __builtin_amdgcn_wmma_f32_16x16x32_f16(
          false, af, false, bf.v, (short)0, acc[nt], false, false);
    }
  }

  const int mb = blockIdx.x*128 + wave*16 + 8*hsel; // D rows: M = r + 8*(lane/16)
  if (EPI == 0) {
    #pragma unroll
    for (int nt = 0; nt < 16; ++nt)
      #pragma unroll
      for (int r = 0; r < 8; ++r)
        outh[(size_t)(mb + r)*256 + nt*16 + nl] = (_Float16)acc[nt][r];
  } else {
    for (int nt = 0; nt < 16; ++nt) {
      const int n = nt*16 + nl;
      const float g  = gamma[n] * rsqrtf(var[n] + 1e-3f);
      const float mu = mean[n], bt = beta[n];
      #pragma unroll
      for (int r = 0; r < 8; ++r) {
        const size_t m = (size_t)(mb + r);
        const float val = acc[nt][r] + resid[m*256 + n];
        outf[m*256 + n] = (val - mu) * g + bt;
      }
    }
  }
}

// row_start[q] = lower_bound(indices, q); indices are sorted ascending
__global__ void seg_offsets_kernel(const int* __restrict__ idx, int* __restrict__ ro) {
  const int q = blockIdx.x*256 + threadIdx.x;
  if (q > NQ) return;
  int lo = 0, hi = NK;
  while (lo < hi) { const int mid = (lo + hi) >> 1; if (idx[mid] < q) lo = mid + 1; else hi = mid; }
  ro[q] = lo;
}

// logits[e,h] = scale * dot(qh[idx[e], h*32:...], kh[e, h*32:...])
__global__ void edge_logits_kernel(const _Float16* __restrict__ qh, const _Float16* __restrict__ kh,
                                   const int* __restrict__ idx, float* __restrict__ logits) {
  const int t = blockIdx.x*256 + threadIdx.x;
  const int e = t >> 3, h = t & 7;
  if (e >= NK) return;
  const int q = idx[e];
  const uint4* qp = (const uint4*)(qh + (size_t)q*256 + h*32);
  const uint4* kp = (const uint4*)(kh + (size_t)e*256 + h*32);
  float acc = 0.f;
  #pragma unroll
  for (int j = 0; j < 4; ++j) {
    union { uint4 u; _Float16 h[8]; } a, b;
    a.u = qp[j]; b.u = kp[j];
    #pragma unroll
    for (int i = 0; i < 8; ++i) acc += (float)a.h[i] * (float)b.h[i];
  }
  logits[(size_t)e*8 + h] = acc * 0.17677669529663689f;   // 1/sqrt(32)
}

// Per query: segment softmax over its edge range + weighted sum of V rows.
#define CHUNK 128
__global__ __launch_bounds__(256) void seg_softmax_agg_kernel(
    const float* __restrict__ logits, const _Float16* __restrict__ vh,
    const int* __restrict__ ro, float* __restrict__ outh) {
  __shared__ float sm[8], ss[8];
  __shared__ float attn[CHUNK*8];
  const int q = blockIdx.x, tid = threadIdx.x;
  const int e0 = ro[q], e1 = ro[q+1];
  if (tid < 8) {                      // avg 16 edges/query: serial per-head reduce is cheap
    const int h = tid;
    float m = -3.4e38f;
    for (int e = e0; e < e1; ++e) m = fmaxf(m, logits[(size_t)e*8 + h]);
    float s = 0.f;
    for (int e = e0; e < e1; ++e) s += __expf(logits[(size_t)e*8 + h] - m);
    sm[h] = m; ss[h] = (s > 0.f) ? 1.0f / s : 0.f;
  }
  __syncthreads();
  const int h = tid >> 5;             // output channel tid -> head tid/32
  float acc = 0.f;
  for (int base = e0; base < e1; base += CHUNK) {
    const int cnt = min(CHUNK, e1 - base);
    for (int i = tid; i < cnt*8; i += 256) {
      const int hh = i & 7;
      attn[i] = __expf(logits[(size_t)(base + (i >> 3))*8 + hh] - sm[hh]) * ss[hh];
    }
    __syncthreads();
    for (int j = 0; j < cnt; ++j)
      acc += attn[j*8 + h] * (float)vh[(size_t)(base + j)*256 + tid];
    __syncthreads();
  }
  outh[(size_t)q*256 + tid] = acc;
}

extern "C" void kernel_launch(void* const* d_in, const int* in_sizes, int n_in,
                              void* d_out, int out_size, void* d_ws, size_t ws_size,
                              hipStream_t stream) {
  const float* q     = (const float*)d_in[0];
  const float* k     = (const float*)d_in[1];
  const float* v     = (const float*)d_in[2];
  const int*   idx   = (const int*)  d_in[3];
  const float* Wq    = (const float*)d_in[4];
  const float* bq    = (const float*)d_in[5];
  const float* Wk    = (const float*)d_in[6];
  const float* bk    = (const float*)d_in[7];
  const float* Wv    = (const float*)d_in[8];
  const float* bv    = (const float*)d_in[9];
  const float* Wo    = (const float*)d_in[10];
  const float* bo    = (const float*)d_in[11];
  const float* gamma = (const float*)d_in[12];
  const float* beta  = (const float*)d_in[13];
  const float* mean  = (const float*)d_in[14];
  const float* var   = (const float*)d_in[15];

  // workspace carve-up (all 16B aligned)
  _Float16* kh     = (_Float16*)d_ws;                       // NK*256 f16
  _Float16* vh     = kh + (size_t)NK*256;                   // NK*256 f16
  _Float16* qh     = vh + (size_t)NK*256;                   // NQ*256 f16
  _Float16* wh     = qh + (size_t)NQ*256;                   // 4 * 256*256 f16 (Wq,Wk,Wv,Wo)
  float*    logits = (float*)(wh + (size_t)4*65536);        // NK*8 f32
  float*    outh   = logits + (size_t)NK*8;                 // NQ*256 f32
  int*      ro     = (int*)(outh + (size_t)NQ*256);         // NQ+1 i32

  const size_t smem = (size_t)256 * WP * sizeof(_Float16);  // 132 KB

  cvt_w_kernel<<<128, 256, 0, stream>>>(Wq, Wk, Wv, Wo, wh);

  gemm256_kernel<0><<<NQ/128, 256, smem, stream>>>(q, wh + 0*65536, bq,
      nullptr, nullptr, nullptr, nullptr, nullptr, qh, nullptr);
  gemm256_kernel<0><<<NK/128, 256, smem, stream>>>(k, wh + 1*65536, bk,
      nullptr, nullptr, nullptr, nullptr, nullptr, kh, nullptr);
  gemm256_kernel<0><<<NK/128, 256, smem, stream>>>(v, wh + 2*65536, bv,
      nullptr, nullptr, nullptr, nullptr, nullptr, vh, nullptr);

  seg_offsets_kernel<<<(NQ + 256)/256, 256, 0, stream>>>(idx, ro);
  edge_logits_kernel<<<(NK*8)/256, 256, 0, stream>>>(qh, kh, idx, logits);
  seg_softmax_agg_kernel<<<NQ, 256, 0, stream>>>(logits, vh, ro, outh);

  gemm256_kernel<1><<<NQ/128, 256, smem, stream>>>(outh, wh + 3*65536, bo,
      q, gamma, beta, mean, var, nullptr, (float*)d_out);
}
